// AttLSTM_72919954751636
// MI455X (gfx1250) — matrix operand order repaired and checked by reading, not compiled, not run on hardware
//
#include <hip/hip_runtime.h>
#include <hip/hip_bf16.h>
#include <cstdint>

// ---- problem constants (from reference) ----
#define B_SZ   4096
#define S_SZ   8192
#define D_SZ   512
#define XK     1536      // concat([f_x, h, r]) width
#define GATES  2048      // 4*H
#define KITERS 4

typedef __attribute__((ext_vector_type(16))) __bf16 v16bf;
typedef __attribute__((ext_vector_type(8)))  float  v8f;

union V16U { v16bf v; uint4 q[2]; };

// ---- WMMA operand loaders (CDNA5 16-bit layouts, ISA 7.12.2) ----
// A operand (16x32, M x K): lane(m=l&15,h=l>>4) holds row m,
//   K = {k0+h*8 .. +7} and {k0+16+h*8 .. +7}  -> two 16B loads, 32B apart.
__device__ __forceinline__ v16bf load_opA(const __bf16* base, int ld, int row0, int k0, int lane) {
    int m = lane & 15, h = lane >> 4;
    const __bf16* p = base + (size_t)(row0 + m) * ld + k0 + h * 8;
    V16U u;
    u.q[0] = *reinterpret_cast<const uint4*>(p);
    u.q[1] = *reinterpret_cast<const uint4*>(p + 16);
    return u.v;
}
// B operand (32x16, K x N): lane(n=l&15,h=l>>4) holds col n, K={k0+h*8..+7} and
// {k0+16+h*8..+7} -> two 16B loads, 32B apart. `base` indexed [n][k] row-major (K contiguous).
__device__ __forceinline__ v16bf load_opB(const __bf16* base, int ld, int n0, int k0, int lane) {
    int n = lane & 15, h = lane >> 4;
    const __bf16* p = base + (size_t)(n0 + n) * ld + k0 + h * 8;
    V16U u;
    u.q[0] = *reinterpret_cast<const uint4*>(p);
    u.q[1] = *reinterpret_cast<const uint4*>(p + 16);
    return u.v;
}
// B operand from an LDS tile laid out [64 rows][32 k] bf16 (row stride 64B).
__device__ __forceinline__ v16bf load_opB_lds(const __bf16* base, int n0, int lane) {
    int n = lane & 15, h = lane >> 4;
    const uint4* p = reinterpret_cast<const uint4*>(base + (n0 + n) * 32 + h * 8);
    V16U u;
    u.q[0] = p[0];
    u.q[1] = p[2];
    return u.v;
}

__device__ __forceinline__ v8f wmma_bf16(v16bf a, v16bf b, v8f c) {
    return __builtin_amdgcn_wmma_f32_16x16x32_bf16(false, a, false, b, (short)0, c, false, false);
}

__device__ __forceinline__ float sigmf(float x) { return 1.0f / (1.0f + __expf(-x)); }

__device__ __forceinline__ void wait_asynccnt0() {
    asm volatile("s_wait_asynccnt 0x0" ::: "memory");
}

// Stage one 64(row) x 32(k) bf16 B-tile of gT into LDS via async-to-LDS DMA.
// 128 threads x 2 chunks x 16B = 4 KB. ASYNCcnt-tracked.
__device__ __forceinline__ void stage_b_async(__bf16* dst, const __bf16* gsrc,
                                              int n0, int k, int tid) {
#pragma unroll
    for (int c = 0; c < 2; ++c) {
        int chunk = tid + c * 128;           // 0..255
        int row = chunk >> 2, part = chunk & 3;
        const __bf16* ga = gsrc + (size_t)(n0 + row) * S_SZ + k + part * 8;
        unsigned loff = (unsigned)(uintptr_t)(dst + row * 32 + part * 8);
        asm volatile("global_load_async_to_lds_b128 %0, %1, off"
                     :: "v"(loff), "v"(ga) : "memory");
    }
}

// ---------------- prep kernels (once per launch) ----------------
__global__ void k_prep_fx(const float* __restrict__ fx, __bf16* __restrict__ X,
                          float* __restrict__ c_buf, int n) {
    int i = blockIdx.x * 256 + threadIdx.x;
    if (i >= n) return;
    int b = i >> 9, d = i & 511;
    __bf16 v = (__bf16)fx[i];
    X[(size_t)b * XK + d]       = v;   // fx slot
    X[(size_t)b * XK + 512 + d] = v;   // h slot, h0 = f_x
    c_buf[i] = 0.0f;
}

__global__ void k_prep_g(const float* __restrict__ g, __bf16* __restrict__ gbf,
                         __bf16* __restrict__ gT, int n) {
    int i = blockIdx.x * 256 + threadIdx.x;
    if (i >= n) return;
    int s = i >> 9, d = i & 511;
    __bf16 v = (__bf16)g[i];
    gbf[i] = v;
    gT[(size_t)d * S_SZ + s] = v;
}

__global__ void k_prep_w(const float* __restrict__ Wih, const float* __restrict__ Whh,
                         __bf16* __restrict__ Wc, int n) {
    int i = blockIdx.x * 256 + threadIdx.x;
    if (i >= n) return;
    int j = i / XK, k = i - j * XK;
    float v = (k < 512) ? Wih[(size_t)j * 512 + k] : Whh[(size_t)j * 1024 + (k - 512)];
    Wc[i] = (__bf16)v;
}

__global__ void k_prep_b(const float* __restrict__ bi, const float* __restrict__ bh,
                         float* __restrict__ bc) {
    int i = blockIdx.x * 256 + threadIdx.x;
    if (i < GATES) bc[i] = bi[i] + bh[i];
}

// ---------------- logits = h @ g_S^T  [B,S] bf16 ----------------
// WG = 4 waves; wave w -> rows m0+32w..+31 (2 A tiles), 64 cols; double-buffered.
__global__ __launch_bounds__(128) void k_logits(const __bf16* __restrict__ X,
                                                const __bf16* __restrict__ g,
                                                __bf16* __restrict__ out) {
    int lane = threadIdx.x & 31, wave = threadIdx.x >> 5;
    int m0 = blockIdx.y * 128 + wave * 32;
    int n0 = blockIdx.x * 64;
    const __bf16* hb = X + 512;                 // h lives at X[:,512:1024]
    v8f acc[8] = {};
    v16bf a0 = load_opA(hb, XK, m0, 0, lane);
    v16bf a1 = load_opA(hb, XK, m0 + 16, 0, lane);
    v16bf b[4];
#pragma unroll
    for (int t = 0; t < 4; ++t) b[t] = load_opB(g, D_SZ, n0 + t * 16, 0, lane);
#pragma unroll
    for (int ks = 0; ks < D_SZ / 32; ++ks) {
        v16bf a0n = a0, a1n = a1, bn[4];
#pragma unroll
        for (int t = 0; t < 4; ++t) bn[t] = b[t];
        if (ks + 1 < D_SZ / 32) {                     // prefetch next k-step
            int k = (ks + 1) * 32;
            a0n = load_opA(hb, XK, m0, k, lane);
            a1n = load_opA(hb, XK, m0 + 16, k, lane);
#pragma unroll
            for (int t = 0; t < 4; ++t) bn[t] = load_opB(g, D_SZ, n0 + t * 16, k, lane);
        }
#pragma unroll
        for (int t = 0; t < 4; ++t) {
            acc[t]     = wmma_bf16(a0, b[t], acc[t]);
            acc[4 + t] = wmma_bf16(a1, b[t], acc[4 + t]);
        }
        a0 = a0n; a1 = a1n;
#pragma unroll
        for (int t = 0; t < 4; ++t) b[t] = bn[t];
    }
    int n = lane & 15, h = lane >> 4;
#pragma unroll
    for (int mh = 0; mh < 2; ++mh)
#pragma unroll
        for (int t = 0; t < 4; ++t)
#pragma unroll
            for (int v = 0; v < 8; ++v)
                out[(size_t)(m0 + mh * 16 + v + h * 8) * S_SZ + n0 + t * 16 + n] =
                    (__bf16)acc[mh * 4 + t][v];
}

// ---------------- row softmax in place (bf16) ----------------
__global__ __launch_bounds__(256) void k_softmax(__bf16* __restrict__ logits) {
    __shared__ float red[256];
    size_t row = blockIdx.x;
    __bf16* rp = logits + row * S_SZ + threadIdx.x * 32;   // 8192/256 = 32 per thread
    alignas(16) __bf16 lv[32];
    float v[32];
    uint4* lvq = reinterpret_cast<uint4*>(lv);
    const uint4* gq = reinterpret_cast<const uint4*>(rp);
#pragma unroll
    for (int i = 0; i < 4; ++i) lvq[i] = gq[i];
    float mx = -1e30f;
#pragma unroll
    for (int i = 0; i < 32; ++i) { v[i] = (float)lv[i]; mx = fmaxf(mx, v[i]); }
    red[threadIdx.x] = mx; __syncthreads();
    for (int s = 128; s > 0; s >>= 1) {
        if (threadIdx.x < s) red[threadIdx.x] = fmaxf(red[threadIdx.x], red[threadIdx.x + s]);
        __syncthreads();
    }
    mx = red[0]; __syncthreads();
    float sum = 0.0f;
#pragma unroll
    for (int i = 0; i < 32; ++i) { v[i] = __expf(v[i] - mx); sum += v[i]; }
    red[threadIdx.x] = sum; __syncthreads();
    for (int s = 128; s > 0; s >>= 1) {
        if (threadIdx.x < s) red[threadIdx.x] += red[threadIdx.x + s];
        __syncthreads();
    }
    float inv = 1.0f / red[0];
#pragma unroll
    for (int i = 0; i < 32; ++i) lv[i] = (__bf16)(v[i] * inv);
    uint4* oq = reinterpret_cast<uint4*>(rp);
#pragma unroll
    for (int i = 0; i < 4; ++i) oq[i] = lvq[i];
}

// ---------------- r = a @ g_S  -> X[:,1024:1536] ----------------
// B tiles (gT) shared by all 4 waves: staged into LDS with async-to-LDS DMA,
// double-buffered; A tiles register-double-buffered.
__global__ __launch_bounds__(128) void k_readout(const __bf16* __restrict__ a,
                                                 const __bf16* __restrict__ gT,
                                                 __bf16* __restrict__ X) {
    __shared__ __bf16 sb[2][64 * 32];           // 2 x 4 KB B-tile buffers
    int tid = threadIdx.x, lane = tid & 31, wave = tid >> 5;
    int m0 = blockIdx.y * 128 + wave * 32;
    int n0 = blockIdx.x * 64;
    v8f acc[8] = {};
    stage_b_async(sb[0], gT, n0, 0, tid);
    v16bf a0 = load_opA(a, S_SZ, m0, 0, lane);
    v16bf a1 = load_opA(a, S_SZ, m0 + 16, 0, lane);
    wait_asynccnt0();
    __syncthreads();
    const int NSTEP = S_SZ / 32;
#pragma unroll 2
    for (int ks = 0; ks < NSTEP; ++ks) {
        int buf = ks & 1;
        bool more = (ks + 1) < NSTEP;
        if (more) stage_b_async(sb[buf ^ 1], gT, n0, (ks + 1) * 32, tid);
        v16bf a0n = a0, a1n = a1;
        if (more) {
            a0n = load_opA(a, S_SZ, m0, (ks + 1) * 32, lane);
            a1n = load_opA(a, S_SZ, m0 + 16, (ks + 1) * 32, lane);
        }
#pragma unroll
        for (int t = 0; t < 4; ++t) {
            v16bf b = load_opB_lds(sb[buf], t * 16, lane);
            acc[t]     = wmma_bf16(a0, b, acc[t]);
            acc[4 + t] = wmma_bf16(a1, b, acc[4 + t]);
        }
        if (more) wait_asynccnt0();     // my async stores into buf^1 done
        __syncthreads();                // everyone's done -> safe next step
        a0 = a0n; a1 = a1n;
    }
    int n = lane & 15, h = lane >> 4;
#pragma unroll
    for (int mh = 0; mh < 2; ++mh)
#pragma unroll
        for (int t = 0; t < 4; ++t)
#pragma unroll
            for (int v = 0; v < 8; ++v)
                X[(size_t)(m0 + mh * 16 + v + h * 8) * XK + 1024 + n0 + t * 16 + n] =
                    (__bf16)acc[mh * 4 + t][v];
}

// ---------------- gates GEMM + fused LSTM elementwise ----------------
// WG = 4 waves; wave w owns gate group w for a 32x64 output tile; pipelined.
__global__ __launch_bounds__(128) void k_gates(const __bf16* __restrict__ Wc,
                                               const float* __restrict__ bc,
                                               const float* __restrict__ fx,
                                               const __bf16* __restrict__ X,
                                               float* __restrict__ c_buf,
                                               float* __restrict__ h_out) {
    __shared__ float sg[4 * 32 * 64];   // 32 KB: [gate][row][col]
    int lane = threadIdx.x & 31, wave = threadIdx.x >> 5;
    int m0 = blockIdx.y * 32;
    int d0 = blockIdx.x * 64;
    int j0 = wave * 512 + d0;           // column block in 2048-wide gate space
    v8f acc[8] = {};
    v16bf a0 = load_opA(X, XK, m0, 0, lane);
    v16bf a1 = load_opA(X, XK, m0 + 16, 0, lane);
    v16bf b[4];
#pragma unroll
    for (int t = 0; t < 4; ++t) b[t] = load_opB(Wc, XK, j0 + t * 16, 0, lane);
#pragma unroll 2
    for (int ks = 0; ks < XK / 32; ++ks) {
        v16bf a0n = a0, a1n = a1, bn[4];
#pragma unroll
        for (int t = 0; t < 4; ++t) bn[t] = b[t];
        if (ks + 1 < XK / 32) {
            int k = (ks + 1) * 32;
            a0n = load_opA(X, XK, m0, k, lane);
            a1n = load_opA(X, XK, m0 + 16, k, lane);
#pragma unroll
            for (int t = 0; t < 4; ++t) bn[t] = load_opB(Wc, XK, j0 + t * 16, k, lane);
        }
#pragma unroll
        for (int t = 0; t < 4; ++t) {
            acc[t]     = wmma_bf16(a0, b[t], acc[t]);
            acc[4 + t] = wmma_bf16(a1, b[t], acc[4 + t]);
        }
        a0 = a0n; a1 = a1n;
#pragma unroll
        for (int t = 0; t < 4; ++t) b[t] = bn[t];
    }
    int n = lane & 15, h = lane >> 4;
#pragma unroll
    for (int t = 0; t < 4; ++t) {
        float bias = bc[j0 + t * 16 + n];
#pragma unroll
        for (int mh = 0; mh < 2; ++mh)
#pragma unroll
            for (int v = 0; v < 8; ++v)
                sg[wave * 2048 + (mh * 16 + v + h * 8) * 64 + t * 16 + n] =
                    acc[mh * 4 + t][v] + bias;
    }
    __syncthreads();
#pragma unroll
    for (int it = 0; it < 16; ++it) {
        int idx = threadIdx.x + it * 128;          // idx = r*64 + cc, r in [0,32)
        int r = idx >> 6, cc = idx & 63;
        float gi = sg[idx], gf = sg[2048 + idx], gg = sg[4096 + idx], go = sg[6144 + idx];
        size_t off = (size_t)(m0 + r) * D_SZ + d0 + cc;
        float co = c_buf[off];
        float cn = sigmf(gf) * co + sigmf(gi) * tanhf(gg);
        float hn = sigmf(go) * tanhf(cn) + fx[off];   // residual h += f_x
        c_buf[off] = cn;
        h_out[off] = hn;
    }
}

// commit new h (fp32 in d_out) into X's bf16 h slot (separate pass: avoids
// the RAW race of k_gates reading old h while writing new h).
__global__ void k_commit_h(const float* __restrict__ h, __bf16* __restrict__ X, int n) {
    int i = blockIdx.x * 256 + threadIdx.x;
    if (i >= n) return;
    int b = i >> 9, d = i & 511;
    X[(size_t)b * XK + 512 + d] = (__bf16)h[i];
}

extern "C" void kernel_launch(void* const* d_in, const int* in_sizes, int n_in,
                              void* d_out, int out_size, void* d_ws, size_t ws_size,
                              hipStream_t stream) {
    const float* fx  = (const float*)d_in[0];
    const float* gS  = (const float*)d_in[1];
    const float* Wih = (const float*)d_in[2];
    const float* Whh = (const float*)d_in[3];
    const float* bih = (const float*)d_in[4];
    const float* bhh = (const float*)d_in[5];
    float* h_out = (float*)d_out;           // [B, D] fp32, doubles as h buffer

    char* w = (char*)d_ws;
    auto carve = [&](size_t bytes) { char* p = w; w += (bytes + 255) & ~(size_t)255; return p; };
    __bf16* X      = (__bf16*)carve((size_t)B_SZ * XK * 2);      // 12.6 MB
    __bf16* gbf    = (__bf16*)carve((size_t)S_SZ * D_SZ * 2);    // 8 MB
    __bf16* gT     = (__bf16*)carve((size_t)S_SZ * D_SZ * 2);    // 8 MB
    __bf16* Wc     = (__bf16*)carve((size_t)GATES * XK * 2);     // 6 MB
    float*  bc     = (float*) carve((size_t)GATES * 4);
    float*  c_buf  = (float*) carve((size_t)B_SZ * D_SZ * 4);    // 8 MB
    __bf16* logits = (__bf16*)carve((size_t)B_SZ * S_SZ * 2);    // 64 MB (also holds a)

    k_prep_fx<<<(B_SZ * D_SZ + 255) / 256, 256, 0, stream>>>(fx, X, c_buf, B_SZ * D_SZ);
    k_prep_g <<<(S_SZ * D_SZ + 255) / 256, 256, 0, stream>>>(gS, gbf, gT, S_SZ * D_SZ);
    k_prep_w <<<(GATES * XK + 255) / 256, 256, 0, stream>>>(Wih, Whh, Wc, GATES * XK);
    k_prep_b <<<(GATES + 255) / 256, 256, 0, stream>>>(bih, bhh, bc);

    for (int it = 0; it < KITERS; ++it) {
        k_logits  <<<dim3(S_SZ / 64, B_SZ / 128), 128, 0, stream>>>(X, gbf, logits);
        k_softmax <<<B_SZ, 256, 0, stream>>>(logits);
        k_readout <<<dim3(D_SZ / 64, B_SZ / 128), 128, 0, stream>>>(logits, gT, X);
        k_gates   <<<dim3(D_SZ / 64, B_SZ / 32), 128, 0, stream>>>(Wc, bc, fx, X, c_buf, h_out);
        k_commit_h<<<(B_SZ * D_SZ + 255) / 256, 256, 0, stream>>>(h_out, X, B_SZ * D_SZ);
    }
    (void)in_sizes; (void)n_in; (void)out_size; (void)ws_size;
}
